// RPGNN_11149735101084
// MI455X (gfx1250) — matrix operand order repaired and checked by reading, not compiled, not run on hardware
//
#include <hip/hip_runtime.h>

// ---------------------------------------------------------------------------
// Problem constants (from reference): NQ=1024, K=16, IN=512, OUT=256, H=4
// NT = 17408 nodes, NE = 16384 query-neighbor pairs.
// GEMMs dominate (~61 GFLOP); run them on v_wmma_f32_16x16x32_bf16.
// Graph structure: neighbor nodes have 2 in-edges (query + self),
// query nodes have 17 in-edges (16 neighbors + self) -> dense softmax.
// ---------------------------------------------------------------------------
#define NQn  1024
#define Kn   16
#define INn  512
#define OUTn 256
#define Hn   4
#define NTn  17408
#define NEn  16384

typedef __attribute__((ext_vector_type(16))) __bf16 v16bf;
typedef __attribute__((ext_vector_type(8)))  float  v8f;

// Async global->LDS staging (CDNA5 GLOBAL_LOAD_ASYNC_TO_LDS_B128, ASYNCcnt).
// Gated on builtin availability; falls back to the synchronous path.
#if defined(__gfx1250__) && __has_builtin(__builtin_amdgcn_global_load_async_to_lds_b128)
  #define USE_ASYNC_LDS 1
#else
  #define USE_ASYNC_LDS 0
#endif

// Builtin parameter types (from hipcc diagnostic): pointer to int4 vector,
// global (AS1 printed as __device__) source / LDS (AS3) destination.
typedef int v4i_vs __attribute__((vector_size(16)));
typedef __attribute__((address_space(1))) v4i_vs glob_v4i;
typedef __attribute__((address_space(3))) v4i_vs lds_v4i;

__device__ __forceinline__ unsigned short f2bf(float f) {
  unsigned int u = __float_as_uint(f);
  u += 0x7fffu + ((u >> 16) & 1u);          // round-to-nearest-even
  return (unsigned short)(u >> 16);
}
__device__ __forceinline__ float lrelu(float x) { return x > 0.f ? x : 0.2f * x; }

// ---------------------------------------------------------------------------
// f32 -> bf16 conversion (weights + activations)
// ---------------------------------------------------------------------------
__global__ void cvt_f32_bf16(const float* __restrict__ src,
                             unsigned short* __restrict__ dst, int n) {
  int i = blockIdx.x * blockDim.x + threadIdx.x;
  if (i < n) dst[i] = f2bf(src[i]);
}

// ---------------------------------------------------------------------------
// bf16 WMMA GEMM:  C[M,N] = A[M,K] * B[K,N]  (+bias, +relu, +bf16 copy)
// Block tile 128x64, BK=32 (one WMMA K-step). 8 waves, each owns 32x32.
// A staged [row][k] in LDS (async b128 when available), B staged transposed
// [n][k] so both fragments are contiguous 8-byte LDS runs matching the ISA
// 16-bit A/B lane layout:
//   lanes 0-15: K 0..7 & 16..23 ; lanes 16-31: K 8..15 & 24..31.
// M % 128 == 0, N % 64 == 0, K % 32 == 0 for all call sites here.
// ---------------------------------------------------------------------------
__global__ __launch_bounds__(256)
void gemm_bf16_wmma(const unsigned short* __restrict__ A,
                    const unsigned short* __restrict__ B,
                    float* __restrict__ C,
                    unsigned short* __restrict__ Cb,   // optional bf16 copy
                    const float* __restrict__ bias,    // optional
                    int M, int N, int K, int relu) {
  constexpr int BM = 128, BN = 64, BK = 32;
  constexpr int LDA = 40;   // halves per A row (32 + 8 pad -> 80B rows)
  constexpr int LDB = 40;   // halves per B(T) row

  __shared__ unsigned short As[BM * LDA];
  __shared__ unsigned short Bs[BN * LDB];

  const int tid  = threadIdx.x;
  const int wid  = tid >> 5;
  const int lane = tid & 31;
  const int bm = blockIdx.y * BM;
  const int bn = blockIdx.x * BN;

  const int waveRow = (wid & 3) * 32;   // 4 row-bands
  const int waveCol = (wid >> 2) * 32;  // 2 col-bands

  const int half = lane >> 4;           // 0 / 1
  const int l16  = lane & 15;
  const int kb   = half * 8;

  v8f zero = {0.f,0.f,0.f,0.f,0.f,0.f,0.f,0.f};
  v8f acc[2][2];
  acc[0][0] = zero; acc[0][1] = zero; acc[1][0] = zero; acc[1][1] = zero;

  for (int k0 = 0; k0 < K; k0 += BK) {
    __syncthreads();
    // ---- stage A tile (128x32): 2 passes of 64 rows, one b128 per thread ---
    #pragma unroll
    for (int p = 0; p < 2; ++p) {
      int rowA = p * 64 + (tid >> 2);
      int kc   = (tid & 3) * 8;
      const unsigned short* g = &A[(size_t)(bm + rowA) * K + k0 + kc];
      unsigned short* l = &As[rowA * LDA + kc];
#if USE_ASYNC_LDS
      // per-lane contiguous 16B global run -> per-lane LDS dest: exact
      // semantics of GLOBAL_LOAD_ASYNC_TO_LDS_B128 (tracked by ASYNCcnt)
      __builtin_amdgcn_global_load_async_to_lds_b128(
          (glob_v4i*)g, (lds_v4i*)l, 0, 0);
#else
      uint4 q = *(const uint4*)g;
      unsigned int* dst = (unsigned int*)l;
      dst[0] = q.x; dst[1] = q.y; dst[2] = q.z; dst[3] = q.w;
#endif
    }
    // ---- stage B tile (32x64) transposed into [n][k] (element transpose
    //      must pass through VGPRs; async copy cannot reshuffle halves) ----
    {
      int kr = tid >> 3;
      int nc = (tid & 7) * 8;
      uint4 q = *(const uint4*)&B[(size_t)(k0 + kr) * N + bn + nc];
      unsigned int w[4] = {q.x, q.y, q.z, q.w};
      #pragma unroll
      for (int j = 0; j < 8; ++j)
        Bs[(nc + j) * LDB + kr] = (unsigned short)(w[j >> 1] >> ((j & 1) * 16));
    }
    // prefetch next K tile while we compute (global_prefetch_b8)
    if (k0 + BK < K) {
      __builtin_prefetch(&A[(size_t)(bm + (tid >> 2)) * K + k0 + BK], 0, 1);
      __builtin_prefetch(&B[(size_t)(k0 + BK + (tid >> 3)) * N + bn], 0, 1);
    }
#if USE_ASYNC_LDS
  #if __has_builtin(__builtin_amdgcn_s_wait_asynccnt)
    __builtin_amdgcn_s_wait_asynccnt(0);
  #else
    asm volatile("s_wait_asynccnt 0x0" ::: "memory");
  #endif
#endif
    __syncthreads();

    // ---- build fragments from LDS (8B-aligned runs) ----
    union Frag { uint2 d[4]; v16bf v; };
    Frag fa[2], fb[2];
    #pragma unroll
    for (int r = 0; r < 2; ++r) {
      const unsigned short* ap = &As[(waveRow + r * 16 + l16) * LDA];
      fa[r].d[0] = *(const uint2*)&ap[kb];
      fa[r].d[1] = *(const uint2*)&ap[kb + 4];
      fa[r].d[2] = *(const uint2*)&ap[kb + 16];
      fa[r].d[3] = *(const uint2*)&ap[kb + 20];
    }
    #pragma unroll
    for (int c = 0; c < 2; ++c) {
      const unsigned short* bp = &Bs[(waveCol + c * 16 + l16) * LDB + kb];
      fb[c].d[0] = *(const uint2*)&bp[0];
      fb[c].d[1] = *(const uint2*)&bp[4];
      fb[c].d[2] = *(const uint2*)&bp[16];
      fb[c].d[3] = *(const uint2*)&bp[20];
    }
    #pragma unroll
    for (int r = 0; r < 2; ++r)
      #pragma unroll
      for (int c = 0; c < 2; ++c)
        acc[r][c] = __builtin_amdgcn_wmma_f32_16x16x32_bf16(
            false, fa[r].v, false, fb[c].v, (short)0, acc[r][c], false, false);
  }

  // ---- epilogue: C layout VGPR v -> M = v + 8*half, N = l16 ----
  #pragma unroll
  for (int r = 0; r < 2; ++r) {
    #pragma unroll
    for (int c = 0; c < 2; ++c) {
      int gc = bn + waveCol + c * 16 + l16;
      float bv = bias ? bias[gc] : 0.f;
      #pragma unroll
      for (int v = 0; v < 8; ++v) {
        int gr = bm + waveRow + r * 16 + half * 8 + v;
        float val = acc[r][c][v] + bv;
        if (relu) val = fmaxf(val, 0.f);
        size_t idx = (size_t)gr * N + gc;
        C[idx] = val;
        if (Cb) Cb[idx] = f2bf(val);
      }
    }
  }
}

// ---------------------------------------------------------------------------
// al_s[i,h] = sum_c XP[i,h,c]*a_src[h,c] ; al_d likewise. One block per node.
// ---------------------------------------------------------------------------
__global__ __launch_bounds__(256)
void rowdot_kernel(const float* __restrict__ XP,
                   const float* __restrict__ a_s, const float* __restrict__ a_d,
                   float* __restrict__ al_s, float* __restrict__ al_d, int Cdim) {
  __shared__ float s1[256], s2[256];
  int node = blockIdx.x, tid = threadIdx.x;
  for (int h = 0; h < Hn; ++h) {
    const float* xp = XP + ((size_t)node * Hn + h) * Cdim;
    float p1 = 0.f, p2 = 0.f;
    for (int c = tid; c < Cdim; c += 256) {
      float x = xp[c];
      p1 += x * a_s[h * Cdim + c];
      p2 += x * a_d[h * Cdim + c];
    }
    s1[tid] = p1; s2[tid] = p2;
    __syncthreads();
    for (int s = 128; s > 0; s >>= 1) {
      if (tid < s) { s1[tid] += s1[tid + s]; s2[tid] += s2[tid + s]; }
      __syncthreads();
    }
    if (tid == 0) { al_s[node * Hn + h] = s1[0]; al_d[node * Hn + h] = s2[0]; }
    __syncthreads();
  }
}

// ---------------------------------------------------------------------------
// GAT aggregation for neighbor nodes: exactly 2 in-edges (query, self).
// out = relu(mean_h(att_q*xp[q] + att_s*xp[self]) + b)
// ---------------------------------------------------------------------------
__global__ __launch_bounds__(128)
void att_nbr_kernel(const float* __restrict__ XP,
                    const float* __restrict__ al_s, const float* __restrict__ al_d,
                    const float* __restrict__ bias,
                    float* __restrict__ Xout, unsigned short* __restrict__ Xoutb,
                    int Cdim) {
  int j = blockIdx.x, node = NQn + j, q = j / Kn;
  float wq[Hn], ws[Hn];
  #pragma unroll
  for (int h = 0; h < Hn; ++h) {
    float aq = lrelu(al_s[q * Hn + h]    + al_d[node * Hn + h]);
    float as = lrelu(al_s[node * Hn + h] + al_d[node * Hn + h]);
    float m  = fmaxf(aq, as);
    float e1 = __expf(aq - m), e2 = __expf(as - m);
    float inv = 1.f / (e1 + e2 + 1e-16f);
    wq[h] = e1 * inv; ws[h] = e2 * inv;
  }
  for (int c = threadIdx.x; c < Cdim; c += 128) {
    float acc = 0.f;
    #pragma unroll
    for (int h = 0; h < Hn; ++h)
      acc += wq[h] * XP[((size_t)q * Hn + h) * Cdim + c] +
             ws[h] * XP[((size_t)node * Hn + h) * Cdim + c];
    float v = fmaxf(acc * (1.f / Hn) + bias[c], 0.f);
    size_t idx = (size_t)node * Cdim + c;
    Xout[idx] = v;
    if (Xoutb) Xoutb[idx] = f2bf(v);
  }
}

// ---------------------------------------------------------------------------
// GAT aggregation for query nodes: 17 in-edges (16 neighbors + self).
// ---------------------------------------------------------------------------
__global__ __launch_bounds__(256)
void att_qry_kernel(const float* __restrict__ XP,
                    const float* __restrict__ al_s, const float* __restrict__ al_d,
                    const float* __restrict__ bias,
                    float* __restrict__ Xout, unsigned short* __restrict__ Xoutb,
                    int Cdim) {
  int i = blockIdx.x;
  for (int c = threadIdx.x; c < Cdim; c += 256) {
    float acc = 0.f;
    for (int h = 0; h < Hn; ++h) {
      float ad = al_d[i * Hn + h];
      float a[Kn + 1], m = -1e30f;
      #pragma unroll
      for (int k = 0; k < Kn; ++k) {
        a[k] = lrelu(al_s[(NQn + i * Kn + k) * Hn + h] + ad);
        m = fmaxf(m, a[k]);
      }
      a[Kn] = lrelu(al_s[i * Hn + h] + ad);
      m = fmaxf(m, a[Kn]);
      float den = 1e-16f;
      #pragma unroll
      for (int k = 0; k <= Kn; ++k) { a[k] = __expf(a[k] - m); den += a[k]; }
      float inv = 1.f / den, sum = 0.f;
      #pragma unroll
      for (int k = 0; k < Kn; ++k)
        sum += a[k] * XP[((size_t)(NQn + i * Kn + k) * Hn + h) * Cdim + c];
      sum += a[Kn] * XP[((size_t)i * Hn + h) * Cdim + c];
      acc += sum * inv;
    }
    float v = fmaxf(acc * (1.f / Hn) + bias[c], 0.f);
    size_t idx = (size_t)i * Cdim + c;
    Xout[idx] = v;
    if (Xoutb) Xoutb[idx] = f2bf(v);
  }
}

// ---------------------------------------------------------------------------
// edge_atts[i,k] = softmax_k( dot(qf[i], nbf[i,k]) ). One block per query.
// 16 groups of 16 lanes, each group does one 256-length dot product.
// ---------------------------------------------------------------------------
__global__ __launch_bounds__(256)
void edgeatt_kernel(const float* __restrict__ X2,
                    float* __restrict__ att_out, float* __restrict__ att_ws) {
  __shared__ float sm[256];
  __shared__ float logits[Kn];
  int i = blockIdx.x, tid = threadIdx.x;
  int g = tid >> 4, l = tid & 15;
  const float* qf = X2 + (size_t)i * OUTn;
  const float* nb = X2 + (size_t)(NQn + i * Kn + g) * OUTn;
  float p = 0.f;
  for (int c = l; c < OUTn; c += 16) p += qf[c] * nb[c];
  sm[tid] = p;
  __syncthreads();
  for (int s = 8; s > 0; s >>= 1) {
    if (l < s) sm[tid] += sm[tid + s];
    __syncthreads();
  }
  if (l == 0) logits[g] = sm[tid];
  __syncthreads();
  if (tid == 0) {
    float m = -1e30f;
    #pragma unroll
    for (int k = 0; k < Kn; ++k) m = fmaxf(m, logits[k]);
    float e[Kn], den = 0.f;
    #pragma unroll
    for (int k = 0; k < Kn; ++k) { e[k] = __expf(logits[k] - m); den += e[k]; }
    float inv = 1.f / den;
    #pragma unroll
    for (int k = 0; k < Kn; ++k) {
      float a = e[k] * inv;
      att_out[i * Kn + k] = a;
      att_ws[i * Kn + k]  = a;
    }
  }
}

// edge_features[e] = concat(qf[e/K], nbf[e]) -> bf16
__global__ __launch_bounds__(256)
void build_ef_kernel(const float* __restrict__ X2, unsigned short* __restrict__ EFb) {
  int e = blockIdx.x, tid = threadIdx.x, q = e / Kn;
  EFb[(size_t)e * (2 * OUTn) + tid]        = f2bf(X2[(size_t)q * OUTn + tid]);
  EFb[(size_t)e * (2 * OUTn) + OUTn + tid] = f2bf(X2[(size_t)(NQn + e) * OUTn + tid]);
}

// rel_pose[r, 0:3] = h @ xr_W + xr_b ; rel_pose[r, 3:7] = h @ qr_W + qr_b
__global__ __launch_bounds__(256)
void heads_kernel(const float* __restrict__ H2,
                  const float* __restrict__ xrW, const float* __restrict__ xrb,
                  const float* __restrict__ qrW, const float* __restrict__ qrb,
                  float* __restrict__ rel) {
  int idx = blockIdx.x * blockDim.x + threadIdx.x;
  if (idx >= NEn * 7) return;
  int r = idx / 7, o = idx % 7;
  const float* h = H2 + (size_t)r * OUTn;
  float acc;
  if (o < 3) {
    acc = xrb[o];
    for (int c = 0; c < OUTn; ++c) acc += h[c] * xrW[c * 3 + o];
  } else {
    int oo = o - 3;
    acc = qrb[oo];
    for (int c = 0; c < OUTn; ++c) acc += h[c] * qrW[c * 4 + oo];
  }
  rel[idx] = acc;
}

// abs_pose: quaternion math + attention-weighted sum + normalize
__global__ __launch_bounds__(128)
void finalize_kernel(const float* __restrict__ rel, const float* __restrict__ att,
                     const float* __restrict__ nposes, float* __restrict__ abs_out) {
  int i = blockIdx.x * blockDim.x + threadIdx.x;
  if (i >= NQn) return;
  float acc[7] = {0.f, 0.f, 0.f, 0.f, 0.f, 0.f, 0.f};
  for (int k = 0; k < Kn; ++k) {
    int e = i * Kn + k;
    const float* rp = rel + (size_t)e * 7;
    const float* np = nposes + (size_t)e * 7;
    float px = np[0] + rp[0], py = np[1] + rp[1], pz = np[2] + rp[2];
    float q1w = rp[3], q1x = rp[4], q1y = rp[5], q1z = rp[6];
    float q2w = np[3], q2x = -np[4], q2y = -np[5], q2z = -np[6];   // qinv
    float s  = q1w * q2w - (q1x * q2x + q1y * q2y + q1z * q2z);
    float vx = q1x * q2w + q2x * q1w + (q1y * q2z - q1z * q2y);
    float vy = q1y * q2w + q2y * q1w + (q1z * q2x - q1x * q2z);
    float vz = q1z * q2w + q2z * q1w + (q1x * q2y - q1y * q2x);
    float a = att[e];
    acc[0] += a * px; acc[1] += a * py; acc[2] += a * pz;
    acc[3] += a * s;  acc[4] += a * vx; acc[5] += a * vy; acc[6] += a * vz;
  }
  float n = sqrtf(acc[3]*acc[3] + acc[4]*acc[4] + acc[5]*acc[5] + acc[6]*acc[6]);
  float inv = 1.f / n;
  abs_out[i * 7 + 0] = acc[0];
  abs_out[i * 7 + 1] = acc[1];
  abs_out[i * 7 + 2] = acc[2];
  abs_out[i * 7 + 3] = acc[3] * inv;
  abs_out[i * 7 + 4] = acc[4] * inv;
  abs_out[i * 7 + 5] = acc[5] * inv;
  abs_out[i * 7 + 6] = acc[6] * inv;
}

// ---------------------------------------------------------------------------
extern "C" void kernel_launch(void* const* d_in, const int* in_sizes, int n_in,
                              void* d_out, int out_size, void* d_ws, size_t ws_size,
                              hipStream_t stream) {
  const float* q_enc  = (const float*)d_in[0];
  const float* n_enc  = (const float*)d_in[1];
  const float* nposes = (const float*)d_in[2];
  const float* gat0_W = (const float*)d_in[3];
  const float* gat0_as= (const float*)d_in[4];
  const float* gat0_ad= (const float*)d_in[5];
  const float* gat0_b = (const float*)d_in[6];
  const float* gat1_W = (const float*)d_in[7];
  const float* gat1_as= (const float*)d_in[8];
  const float* gat1_ad= (const float*)d_in[9];
  const float* gat1_b = (const float*)d_in[10];
  const float* ef_W1  = (const float*)d_in[11];
  const float* ef_b1  = (const float*)d_in[12];
  const float* ef_W2  = (const float*)d_in[13];
  const float* ef_b2  = (const float*)d_in[14];
  const float* xr_W   = (const float*)d_in[15];
  const float* xr_b   = (const float*)d_in[16];
  const float* qr_W   = (const float*)d_in[17];
  const float* qr_b   = (const float*)d_in[18];

  float* out    = (float*)d_out;
  float* relOut = out;                       // 16384*7 = 114688
  float* attOut = out + (size_t)NEn * 7;     // 16384
  float* absOut = attOut + (size_t)NQn * Kn; // 1024*7

  char* ws = (char*)d_ws;
  size_t off = 0;
  auto alloc = [&](size_t bytes) -> char* {
    char* p = ws + off;
    off += (bytes + 255) & ~(size_t)255;
    return p;
  };
  unsigned short* W0b  = (unsigned short*)alloc((size_t)INn * (Hn*INn)  * 2);
  unsigned short* W1b  = (unsigned short*)alloc((size_t)INn * (Hn*OUTn) * 2);
  unsigned short* We1b = (unsigned short*)alloc((size_t)(2*OUTn) * OUTn * 2);
  unsigned short* We2b = (unsigned short*)alloc((size_t)OUTn * OUTn * 2);
  unsigned short* X0b  = (unsigned short*)alloc((size_t)NTn * INn * 2);
  float* XP  = (float*)alloc((size_t)NTn * (Hn*INn) * 4);   // reused (see below)
  float* alS = (float*)alloc((size_t)NTn * Hn * 4);
  float* alD = (float*)alloc((size_t)NTn * Hn * 4);
  float* X1  = (float*)alloc((size_t)NTn * INn * 4);
  unsigned short* X1b = (unsigned short*)alloc((size_t)NTn * INn * 2);
  float* X2  = (float*)alloc((size_t)NTn * OUTn * 4);
  float* attw= (float*)alloc((size_t)NQn * Kn * 4);
  // sub-allocations inside the big XP region once layer-1 projections are done
  float* H1            = XP;
  unsigned short* H1b  = (unsigned short*)((char*)XP + (size_t)NEn * OUTn * 4);
  float* H2            = (float*)((char*)H1b + (size_t)NEn * OUTn * 2);
  unsigned short* EFb  = X1b;   // reuse after GEMM2 consumed X1b

  const int T = 256;
  auto cdiv = [](int a, int b) { return (a + b - 1) / b; };

  // --- bf16 conversions: weights + node features (concat query||neighbor) ---
  cvt_f32_bf16<<<cdiv(INn*Hn*INn, T), T, 0, stream>>>(gat0_W, W0b, INn*Hn*INn);
  cvt_f32_bf16<<<cdiv(INn*Hn*OUTn, T), T, 0, stream>>>(gat1_W, W1b, INn*Hn*OUTn);
  cvt_f32_bf16<<<cdiv(2*OUTn*OUTn, T), T, 0, stream>>>(ef_W1, We1b, 2*OUTn*OUTn);
  cvt_f32_bf16<<<cdiv(OUTn*OUTn, T), T, 0, stream>>>(ef_W2, We2b, OUTn*OUTn);
  cvt_f32_bf16<<<cdiv(NQn*INn, T), T, 0, stream>>>(q_enc, X0b, NQn*INn);
  cvt_f32_bf16<<<cdiv(NEn*INn, T), T, 0, stream>>>(n_enc, X0b + (size_t)NQn*INn, NEn*INn);

  // --- GAT layer 0: XP0 = X0 @ W0 (17408x512x2048, bf16 WMMA) ---
  gemm_bf16_wmma<<<dim3((Hn*INn)/64, NTn/128), 256, 0, stream>>>(
      X0b, W0b, XP, nullptr, nullptr, NTn, Hn*INn, INn, 0);
  rowdot_kernel<<<NTn, 256, 0, stream>>>(XP, gat0_as, gat0_ad, alS, alD, INn);
  att_nbr_kernel<<<NEn, 128, 0, stream>>>(XP, alS, alD, gat0_b, X1, X1b, INn);
  att_qry_kernel<<<NQn, 256, 0, stream>>>(XP, alS, alD, gat0_b, X1, X1b, INn);

  // --- GAT layer 1: XP1 = X1 @ W1 (17408x512x1024) ---
  gemm_bf16_wmma<<<dim3((Hn*OUTn)/64, NTn/128), 256, 0, stream>>>(
      X1b, W1b, XP, nullptr, nullptr, NTn, Hn*OUTn, INn, 0);
  rowdot_kernel<<<NTn, 256, 0, stream>>>(XP, gat1_as, gat1_ad, alS, alD, OUTn);
  att_nbr_kernel<<<NEn, 128, 0, stream>>>(XP, alS, alD, gat1_b, X2, nullptr, OUTn);
  att_qry_kernel<<<NQn, 256, 0, stream>>>(XP, alS, alD, gat1_b, X2, nullptr, OUTn);

  // --- edge attention + edge features ---
  edgeatt_kernel<<<NQn, 256, 0, stream>>>(X2, attOut, attw);
  build_ef_kernel<<<NEn, 256, 0, stream>>>(X2, EFb);

  // --- edge MLP: two fused bias+relu WMMA GEMMs ---
  gemm_bf16_wmma<<<dim3(OUTn/64, NEn/128), 256, 0, stream>>>(
      EFb, We1b, H1, H1b, ef_b1, NEn, OUTn, 2*OUTn, 1);
  gemm_bf16_wmma<<<dim3(OUTn/64, NEn/128), 256, 0, stream>>>(
      H1b, We2b, H2, nullptr, ef_b2, NEn, OUTn, OUTn, 1);

  // --- pose heads + final weighted pose ---
  heads_kernel<<<cdiv(NEn*7, T), T, 0, stream>>>(H2, xr_W, xr_b, qr_W, qr_b, relOut);
  finalize_kernel<<<cdiv(NQn, 128), 128, 0, stream>>>(relOut, attw, nposes, absOut);
}
